// FedKDTeacherModel_61521111547947
// MI455X (gfx1250) — matrix operand order repaired and checked by reading, not compiled, not run on hardware
//
#include <hip/hip_runtime.h>
#include <hip/hip_bf16.h>

// ---------------------------------------------------------------------------
// GAT graph classifier forward on MI455X (gfx1250, wave32, WMMA).
//
// The e2@We [1M x 128 x 128] GEMM per layer folds algebraically into a
// per-layer [3x8] matrix + [8] bias (edge embedding is affine in edge_attr and
// only consumed through the att_edge dot product), plus once-per-call per-node
// sums S[n,3] / in-degree cnt[n] for the mean-filled self-loops. Remaining
// cost: bandwidth-bound gather/scatter (~4.5 GB => ~0.2 ms @ 23.3 TB/s) and
// ~17 GMAC of dense GEMMs on v_wmma_f32_16x16x32_f16 (fp16 in, fp32 acc).
//
// GEMM fragments are fetched as 2x ds_load_b128 each (A row-major staged, B
// staged transposed). B fragments are preloaded in batches of up to 8 tiles
// so WMMAs issue back-to-back behind a single dscnt wait; global staging is
// split into load-clause + convert/store phases.
// ---------------------------------------------------------------------------

typedef __attribute__((ext_vector_type(16))) _Float16 v16h;
typedef __attribute__((ext_vector_type(8)))  _Float16 v8h;
typedef __attribute__((ext_vector_type(4)))  _Float16 v4h;
typedef __attribute__((ext_vector_type(8)))  float    v8f;

#define TB 256

static inline int nblk(long long n, int b) { return (int)((n + b - 1) / b); }

// ---- order-preserving float <-> uint encoding for atomic max ---------------
__device__ __forceinline__ unsigned fenc(float f) {
  unsigned b = __float_as_uint(f);
  return (b & 0x80000000u) ? ~b : (b | 0x80000000u);
}
__device__ __forceinline__ float fdec(unsigned k) {
  unsigned b = (k & 0x80000000u) ? (k & 0x7FFFFFFFu) : ~k;
  return __uint_as_float(b);
}

// ---- fills -----------------------------------------------------------------
__global__ void fill_f32(float* p, float v, long long n) {
  long long i = (long long)blockIdx.x * blockDim.x + threadIdx.x;
  if (i < n) p[i] = v;
}
__global__ void fill_u32(unsigned* p, unsigned v, long long n) {
  long long i = (long long)blockIdx.x * blockDim.x + threadIdx.x;
  if (i < n) p[i] = v;
}

// ---- node embedding: h0 = x @ W[9,128] + b ---------------------------------
__global__ void node_emb_kernel(const float* __restrict__ x, const float* __restrict__ W,
                                const float* __restrict__ b, float* __restrict__ h, int n) {
  long long i = (long long)blockIdx.x * blockDim.x + threadIdx.x;
  if (i >= (long long)n * 128) return;
  int nn = (int)(i >> 7), d = (int)(i & 127);
  float acc = b[d];
#pragma unroll
  for (int r = 0; r < 9; ++r) acc += x[nn * 9 + r] * W[r * 128 + d];
  h[i] = acc;
}

// ---- per-node edge_attr sums + in-degree (once per call) -------------------
__global__ void edge_stats_kernel(const float* __restrict__ ea, const int* __restrict__ ei,
                                  float* __restrict__ S, float* __restrict__ cnt, int E_) {
  int e = blockIdx.x * blockDim.x + threadIdx.x;
  if (e >= E_) return;
  int d = ei[E_ + e];
  atomicAdd(&S[d * 3 + 0], ea[e * 3 + 0]);
  atomicAdd(&S[d * 3 + 1], ea[e * 3 + 1]);
  atomicAdd(&S[d * 3 + 2], ea[e * 3 + 2]);
  atomicAdd(&cnt[d], 1.0f);
}

// ---- fold We & att_edge & edge embedding into [3x8] + [8] per layer --------
__global__ void layer_mats_kernel(const float* We0, const float* We1, const float* We2, const float* We3,
                                  const float* ae0, const float* ae1, const float* ae2, const float* ae3,
                                  const float* __restrict__ eW, const float* __restrict__ eb,
                                  float* __restrict__ M3, float* __restrict__ b8) {
  int l = blockIdx.x;
  const float* We = (l == 0) ? We0 : (l == 1) ? We1 : (l == 2) ? We2 : We3;
  const float* ae = (l == 0) ? ae0 : (l == 1) ? ae1 : (l == 2) ? ae2 : ae3;
  __shared__ float watt[128 * 8];
  for (int idx = threadIdx.x; idx < 1024; idx += blockDim.x) {
    int d = idx >> 3, h = idx & 7;
    float s = 0.f;
#pragma unroll
    for (int c = 0; c < 16; ++c) s += We[d * 128 + h * 16 + c] * ae[h * 16 + c];
    watt[idx] = s;
  }
  __syncthreads();
  if (threadIdx.x < 24) {
    int r = threadIdx.x >> 3, h = threadIdx.x & 7;
    float s = 0.f;
    for (int d = 0; d < 128; ++d) s += eW[r * 128 + d] * watt[d * 8 + h];
    M3[l * 24 + threadIdx.x] = s;
  } else if (threadIdx.x >= 32 && threadIdx.x < 40) {
    int h = threadIdx.x - 32;
    float s = 0.f;
    for (int d = 0; d < 128; ++d) s += eb[d] * watt[d * 8 + h];
    b8[l * 8 + h] = s;
  }
}

// ---- WMMA GEMM: C[M,NC] = act(A[M,K] @ B[K,NC] + bias) ---------------------
// Block: 128 rows x NC cols, 8 wave32 waves (16 rows each). K in 64-chunks.
// A staged row-major f16 (128 x 64); B staged TRANSPOSED f16 (NC x 64) so
// every fragment is two 16-byte ds_load_b128 per CDNA5 ISA 7.12.2 layouts.
template <int NC, bool RELU, bool BIAS>
__launch_bounds__(256)
__global__ void wmma_gemm_kernel(const float* __restrict__ A, const float* __restrict__ B,
                                 const float* __restrict__ bias, float* __restrict__ C,
                                 int M, int K) {
  constexpr int KC = 64;
  constexpr int AST = KC + 8;   // 72 halves = 144 B row stride (16B-aligned)
  constexpr int BST = KC + 8;
  constexpr int NT = NC / 16;
  constexpr int PB = (NT < 8) ? NT : 8;           // B-fragment preload batch
  constexpr int AIT = (128 * KC / 4) / TB;        // = 8 float4 per thread
  constexpr int BIT = (KC * NC / 4) / TB;         // float4 per thread for B
  constexpr int BB = (BIT < 8) ? BIT : 8;         // B staging load batch
  __shared__ _Float16 sA[128 * AST];
  __shared__ _Float16 sBt[NC * BST];
  const int t = threadIdx.x;
  const int wave = t >> 5;
  const int lane = t & 31;
  const int r16 = lane & 15;
  const int hi = lane >> 4;
  const int row0 = blockIdx.x * 128;

  v8f acc[NT] = {};

  for (int kc = 0; kc < K; kc += KC) {
    // ---- stage A: load clause of float4, then cvt + 8B LDS stores ----
    {
      float4 av[AIT];
#pragma unroll
      for (int i = 0; i < AIT; ++i) {
        int idx = t + i * TB;
        int r = idx >> 4;              // / (KC/4)
        int c4 = idx & 15;             // % (KC/4)
        int grow = row0 + r;
        av[i] = make_float4(0.f, 0.f, 0.f, 0.f);
        if (grow < M) av[i] = *(const float4*)(A + (size_t)grow * K + kc + c4 * 4);
      }
#pragma unroll
      for (int i = 0; i < AIT; ++i) {
        int idx = t + i * TB;
        int r = idx >> 4;
        int c4 = idx & 15;
        v4h hv;
        hv[0] = (_Float16)av[i].x; hv[1] = (_Float16)av[i].y;
        hv[2] = (_Float16)av[i].z; hv[3] = (_Float16)av[i].w;
        *(v4h*)&sA[r * AST + c4 * 4] = hv;
      }
    }
    // ---- stage B transposed: batched load clause, then scatter b16 ----
#pragma unroll
    for (int ib = 0; ib < BIT; ib += BB) {
      float4 bv[BB];
#pragma unroll
      for (int j = 0; j < BB; ++j) {
        int idx = t + (ib + j) * TB;
        int c4 = idx % (NC / 4);
        int r = idx / (NC / 4);
        bv[j] = *(const float4*)(B + (size_t)(kc + r) * NC + c4 * 4);
      }
#pragma unroll
      for (int j = 0; j < BB; ++j) {
        int idx = t + (ib + j) * TB;
        int c4 = idx % (NC / 4);
        int r = idx / (NC / 4);
        sBt[(c4 * 4 + 0) * BST + r] = (_Float16)bv[j].x;
        sBt[(c4 * 4 + 1) * BST + r] = (_Float16)bv[j].y;
        sBt[(c4 * 4 + 2) * BST + r] = (_Float16)bv[j].z;
        sBt[(c4 * 4 + 3) * BST + r] = (_Float16)bv[j].w;
      }
    }
    __syncthreads();
#pragma unroll
    for (int kk = 0; kk < KC; kk += 32) {
      // A fragment: lane (r16,hi): K = kk + hi*8 + i (i<8), kk+16+hi*8+(i-8)
      const _Float16* ap = &sA[(wave * 16 + r16) * AST + kk + hi * 8];
      v8h a0 = *(const v8h*)(ap);
      v8h a1 = *(const v8h*)(ap + 16);
      v16h afrag = __builtin_shufflevector(a0, a1, 0, 1, 2, 3, 4, 5, 6, 7,
                                           8, 9, 10, 11, 12, 13, 14, 15);
      // B fragments: preload a batch of PB tiles, then issue PB WMMAs
#pragma unroll
      for (int ntb = 0; ntb < NT; ntb += PB) {
        v16h bf[PB];
#pragma unroll
        for (int j = 0; j < PB; ++j) {
          const _Float16* bp = &sBt[((ntb + j) * 16 + r16) * BST + kk + hi * 8];
          v8h b0 = *(const v8h*)(bp);
          v8h b1 = *(const v8h*)(bp + 16);
          bf[j] = __builtin_shufflevector(b0, b1, 0, 1, 2, 3, 4, 5, 6, 7,
                                          8, 9, 10, 11, 12, 13, 14, 15);
        }
#pragma unroll
        for (int j = 0; j < PB; ++j) {
          acc[ntb + j] = __builtin_amdgcn_wmma_f32_16x16x32_f16(
              false, afrag, false, bf[j], (short)0, acc[ntb + j], false, false);
        }
      }
    }
    __syncthreads();
  }
#pragma unroll
  for (int nt = 0; nt < NT; ++nt) {
    int col = nt * 16 + r16;
    float bv = 0.0f;
    if constexpr (BIAS) bv = bias[col];
#pragma unroll
    for (int i = 0; i < 8; ++i) {
      int grow = row0 + wave * 16 + hi * 8 + i;
      if (grow < M) {
        float v = acc[nt][i] + bv;
        if constexpr (RELU) v = v > 0.f ? v : 0.f;
        C[(size_t)grow * NC + col] = v;
      }
    }
  }
}

// ---- per-node attention dot products (xs . att_src/att_dst) ----------------
__global__ void att_reduce_kernel(const float* __restrict__ xs, const float* __restrict__ asrc,
                                  const float* __restrict__ adst, float* __restrict__ asn,
                                  float* __restrict__ adn, int n) {
  long long i = (long long)blockIdx.x * blockDim.x + threadIdx.x;
  if (i >= (long long)n * 8) return;
  int nn = (int)(i >> 3), h = (int)(i & 7);
  float s1 = 0.f, s2 = 0.f;
#pragma unroll
  for (int c = 0; c < 16; ++c) {
    float v = xs[(size_t)nn * 128 + h * 16 + c];
    s1 += v * asrc[h * 16 + c];
    s2 += v * adst[h * 16 + c];
  }
  asn[i] = s1;
  adn[i] = s2;
}

// ---- alpha = lrelu(src+dst+edge), atomic segment max over dst --------------
__global__ void alpha_max_kernel(const float* __restrict__ ea, const int* __restrict__ ei,
                                 const float* __restrict__ S, const float* __restrict__ cnt,
                                 const float* __restrict__ M3, const float* __restrict__ b8,
                                 const float* __restrict__ asn, const float* __restrict__ adn,
                                 float* __restrict__ alpha, unsigned* __restrict__ menc,
                                 int E_, int N_) {
  int idx = blockIdx.x * blockDim.x + threadIdx.x;
  if (idx >= E_ + N_) return;
  int s, d;
  float ae[8];
  if (idx < E_) {
    s = ei[idx]; d = ei[E_ + idx];
    float a0 = ea[idx * 3], a1 = ea[idx * 3 + 1], a2 = ea[idx * 3 + 2];
#pragma unroll
    for (int h = 0; h < 8; ++h) ae[h] = a0 * M3[h] + a1 * M3[8 + h] + a2 * M3[16 + h] + b8[h];
  } else {
    int n = idx - E_; s = n; d = n;
    float c = cnt[n];
    float inv = 1.0f / fmaxf(c, 1.0f);
    float a0 = S[n * 3], a1 = S[n * 3 + 1], a2 = S[n * 3 + 2];
#pragma unroll
    for (int h = 0; h < 8; ++h)
      ae[h] = (a0 * M3[h] + a1 * M3[8 + h] + a2 * M3[16 + h] + c * b8[h]) * inv;
  }
#pragma unroll
  for (int h = 0; h < 8; ++h) {
    float v = asn[(size_t)s * 8 + h] + adn[(size_t)d * 8 + h] + ae[h];
    v = v > 0.f ? v : 0.2f * v;
    alpha[(size_t)idx * 8 + h] = v;
    atomicMax(&menc[(size_t)d * 8 + h], fenc(v));
  }
}

// ---- a = exp(alpha - m[dst]); atomic segment sum of a ----------------------
__global__ void alpha_exp_kernel(const int* __restrict__ ei, float* __restrict__ alpha,
                                 const unsigned* __restrict__ menc, float* __restrict__ denom,
                                 int E_, int N_) {
  int idx = blockIdx.x * blockDim.x + threadIdx.x;
  if (idx >= E_ + N_) return;
  int d = (idx < E_) ? ei[E_ + idx] : (idx - E_);
#pragma unroll
  for (int h = 0; h < 8; ++h) {
    float m = fdec(menc[(size_t)d * 8 + h]);
    float a = __expf(alpha[(size_t)idx * 8 + h] - m);
    alpha[(size_t)idx * 8 + h] = a;
    atomicAdd(&denom[(size_t)d * 8 + h], a);
  }
}

// ---- out[dst] += xs[src] * a/denom : 32 lanes per edge, float4 gather ------
__global__ void scatter_kernel(const int* __restrict__ ei, const float* __restrict__ alpha,
                               const float* __restrict__ denom, const float* __restrict__ xs,
                               float* __restrict__ out, int E_, int N_) {
  long long gid = (long long)blockIdx.x * blockDim.x + threadIdx.x;
  int idx = (int)(gid >> 5);
  int lane = (int)(gid & 31);
  if (idx >= E_ + N_) return;
  int s, d;
  if (idx < E_) { s = ei[idx]; d = ei[E_ + idx]; } else { s = idx - E_; d = s; }
  int h = lane >> 2;
  float w = alpha[(size_t)idx * 8 + h] / (denom[(size_t)d * 8 + h] + 1e-16f);
  const float4 v = *(const float4*)(xs + (size_t)s * 128 + lane * 4);
  float* o = out + (size_t)d * 128 + lane * 4;
  atomicAdd(o + 0, v.x * w);
  atomicAdd(o + 1, v.y * w);
  atomicAdd(o + 2, v.z * w);
  atomicAdd(o + 3, v.w * w);
}

// ---- +bias, eval BatchNorm, relu, optional residual ------------------------
__global__ void bn_relu_kernel(float* __restrict__ out, const float* __restrict__ b,
                               const float* __restrict__ g, const float* __restrict__ bb,
                               const float* __restrict__ rm, const float* __restrict__ rv,
                               const float* __restrict__ res, int n) {
  long long i = (long long)blockIdx.x * blockDim.x + threadIdx.x;
  if (i >= (long long)n * 128) return;
  int d = (int)(i & 127);
  float v = out[i] + b[d];
  v = (v - rm[d]) * rsqrtf(rv[d] + 1e-5f) * g[d] + bb[d];
  v = v > 0.f ? v : 0.f;
  if (res) v += res[i];
  out[i] = v;
}

// ---- row . weight vector (+bias, optional sigmoid) -------------------------
template <int KD, bool SIG>
__global__ void rowdot_kernel(const float* __restrict__ A, const float* __restrict__ w,
                              const float* __restrict__ b2, float* __restrict__ out, int M) {
  int i = blockIdx.x * blockDim.x + threadIdx.x;
  if (i >= M) return;
  float s = b2[0];
#pragma unroll
  for (int j = 0; j < KD; ++j) s += A[(size_t)i * KD + j] * w[j];
  if constexpr (SIG) s = 1.0f / (1.0f + __expf(-s));
  out[i] = s;
}

// ---- pooling ---------------------------------------------------------------
__global__ void pool_acc_kernel(const float* __restrict__ h, const float* __restrict__ gate,
                                const int* __restrict__ batch, float* __restrict__ gsum,
                                unsigned* __restrict__ gmax, float* __restrict__ gcnt,
                                unsigned* __restrict__ gm, int n) {
  long long gid = (long long)blockIdx.x * blockDim.x + threadIdx.x;
  int nn = (int)(gid >> 5);
  int lane = (int)(gid & 31);
  if (nn >= n) return;
  int b = batch[nn];
#pragma unroll
  for (int j = 0; j < 4; ++j) {
    int dd = lane * 4 + j;
    float v = h[(size_t)nn * 128 + dd];
    atomicAdd(&gsum[(size_t)b * 128 + dd], v);
    atomicMax(&gmax[(size_t)b * 128 + dd], fenc(v));
  }
  if (lane == 0) {
    atomicAdd(&gcnt[b], 1.0f);
    atomicMax(&gm[b], fenc(gate[nn]));
  }
}
__global__ void pool_exp_kernel(const float* __restrict__ gate, const int* __restrict__ batch,
                                const unsigned* __restrict__ gm, float* __restrict__ ga,
                                float* __restrict__ gden, int n) {
  int i = blockIdx.x * blockDim.x + threadIdx.x;
  if (i >= n) return;
  int b = batch[i];
  float a = __expf(gate[i] - fdec(gm[b]));
  ga[i] = a;
  atomicAdd(&gden[b], a);
}
__global__ void pool_att_kernel(const float* __restrict__ h, const float* __restrict__ ga,
                                const float* __restrict__ gden, const int* __restrict__ batch,
                                float* __restrict__ gatt, int n) {
  long long gid = (long long)blockIdx.x * blockDim.x + threadIdx.x;
  int nn = (int)(gid >> 5);
  int lane = (int)(gid & 31);
  if (nn >= n) return;
  int b = batch[nn];
  float w = ga[nn] / (gden[b] + 1e-16f);
#pragma unroll
  for (int j = 0; j < 4; ++j) {
    int dd = lane * 4 + j;
    atomicAdd(&gatt[(size_t)b * 128 + dd], h[(size_t)nn * 128 + dd] * w);
  }
}
__global__ void gfeat_kernel(const float* __restrict__ gsum, const unsigned* __restrict__ gmax,
                             const float* __restrict__ gatt, const float* __restrict__ gcnt,
                             float* __restrict__ gf, int G_) {
  long long i = (long long)blockIdx.x * blockDim.x + threadIdx.x;
  if (i >= (long long)G_ * 384) return;
  int g = (int)(i / 384), k = (int)(i % 384);
  float v;
  if (k < 128)      v = gsum[(size_t)g * 128 + k] / fmaxf(gcnt[g], 1.0f);
  else if (k < 256) v = fdec(gmax[(size_t)g * 128 + (k - 128)]);
  else              v = gatt[(size_t)g * 128 + (k - 256)];
  gf[i] = v;
}

// ---------------------------------------------------------------------------
extern "C" void kernel_launch(void* const* d_in, const int* in_sizes, int n_in,
                              void* d_out, int out_size, void* d_ws, size_t ws_size,
                              hipStream_t stream) {
  const float* x          = (const float*)d_in[0];
  const float* edge_attr  = (const float*)d_in[1];
  const int*   edge_index = (const int*)d_in[2];
  const int*   batch      = (const int*)d_in[3];
  const float* nW = (const float*)d_in[4];
  const float* nb = (const float*)d_in[5];
  const float* eW = (const float*)d_in[6];
  const float* eb = (const float*)d_in[7];
  auto gat = [&](int l, int j) { return (const float*)d_in[8 + l * 6 + j]; };   // W,We,att_src,att_dst,att_edge,b
  auto bnp = [&](int l, int j) { return (const float*)d_in[32 + l * 4 + j]; };  // g,b,rm,rv
  const float* gW1 = (const float*)d_in[48];
  const float* gb1 = (const float*)d_in[49];
  const float* gW2 = (const float*)d_in[50];
  const float* gb2 = (const float*)d_in[51];
  const float* mW1 = (const float*)d_in[52];
  const float* mb1 = (const float*)d_in[53];
  const float* mW2 = (const float*)d_in[54];
  const float* mb2 = (const float*)d_in[55];
  const float* fW1 = (const float*)d_in[56];
  const float* fb1 = (const float*)d_in[57];
  const float* fW2 = (const float*)d_in[58];
  const float* fb2 = (const float*)d_in[59];
  const float* cW1 = (const float*)d_in[60];
  const float* cb1 = (const float*)d_in[61];
  const float* cW2 = (const float*)d_in[62];
  const float* cb2 = (const float*)d_in[63];

  const int N = in_sizes[3];
  const int E = in_sizes[1] / 3;
  const int G = out_size;

  // ---- workspace carving (~450 MB in fp32) ----
  float* base = (float*)d_ws;
  size_t off = 0;
  auto alloc = [&](size_t n) { float* r = base + off; off += n; return r; };
  float*    h0    = alloc((size_t)N * 128);
  float*    bufA  = alloc((size_t)N * 128);
  float*    bufB  = alloc((size_t)N * 128);
  float*    xs    = alloc((size_t)N * 128);       // pooling arena aliases here later
  float*    alpha = alloc((size_t)(E + N) * 8);
  float*    asn   = alloc((size_t)N * 8);
  float*    adn   = alloc((size_t)N * 8);
  unsigned* menc  = (unsigned*)alloc((size_t)N * 8);
  float*    denom = alloc((size_t)N * 8);
  float*    S     = alloc((size_t)N * 3);
  float*    cnt   = alloc((size_t)N);
  float*    M3    = alloc(4 * 24);
  float*    b8    = alloc(4 * 8);
  (void)ws_size; (void)n_in;

  // ---- once-per-call precomputation ----
  fill_f32<<<nblk((long long)N * 3, TB), TB, 0, stream>>>(S, 0.f, (long long)N * 3);
  fill_f32<<<nblk(N, TB), TB, 0, stream>>>(cnt, 0.f, N);
  edge_stats_kernel<<<nblk(E, TB), TB, 0, stream>>>(edge_attr, edge_index, S, cnt, E);
  layer_mats_kernel<<<4, 256, 0, stream>>>(gat(0, 1), gat(1, 1), gat(2, 1), gat(3, 1),
                                           gat(0, 4), gat(1, 4), gat(2, 4), gat(3, 4),
                                           eW, eb, M3, b8);
  node_emb_kernel<<<nblk((long long)N * 128, TB), TB, 0, stream>>>(x, nW, nb, h0, N);

  // ---- 4 GAT layers ----
  float* hcur = h0;
  float* dsts[4] = {bufA, bufB, bufA, bufB};
  for (int l = 0; l < 4; ++l) {
    float* hdst = dsts[l];
    wmma_gemm_kernel<128, false, false><<<nblk(N, 128), 256, 0, stream>>>(
        hcur, gat(l, 0), nullptr, xs, N, 128);
    att_reduce_kernel<<<nblk((long long)N * 8, TB), TB, 0, stream>>>(
        xs, gat(l, 2), gat(l, 3), asn, adn, N);
    fill_u32<<<nblk((long long)N * 8, TB), TB, 0, stream>>>(menc, 0u, (long long)N * 8);
    fill_f32<<<nblk((long long)N * 8, TB), TB, 0, stream>>>(denom, 0.f, (long long)N * 8);
    alpha_max_kernel<<<nblk(E + N, TB), TB, 0, stream>>>(
        edge_attr, edge_index, S, cnt, M3 + l * 24, b8 + l * 8, asn, adn, alpha, menc, E, N);
    alpha_exp_kernel<<<nblk(E + N, TB), TB, 0, stream>>>(edge_index, alpha, menc, denom, E, N);
    fill_f32<<<nblk((long long)N * 128, TB), TB, 0, stream>>>(hdst, 0.f, (long long)N * 128);
    scatter_kernel<<<nblk((long long)(E + N) * 32, TB), TB, 0, stream>>>(
        edge_index, alpha, denom, xs, hdst, E, N);
    bn_relu_kernel<<<nblk((long long)N * 128, TB), TB, 0, stream>>>(
        hdst, gat(l, 5), bnp(l, 0), bnp(l, 1), bnp(l, 2), bnp(l, 3),
        (l == 2) ? h0 : nullptr, N);
    hcur = hdst;
  }

  // ---- pooling arena aliased into (now dead) xs buffer ----
  size_t poff = 0;
  auto palloc = [&](size_t n) { float* r = xs + poff; poff += n; return r; };
  float*    g1    = palloc((size_t)N * 64);
  float*    gatev = palloc((size_t)N);
  float*    ga    = palloc((size_t)N);
  float*    gsum  = palloc((size_t)G * 128);
  unsigned* gmax  = (unsigned*)palloc((size_t)G * 128);
  float*    gatt  = palloc((size_t)G * 128);
  float*    gcnt  = palloc((size_t)G);
  unsigned* gm    = (unsigned*)palloc((size_t)G);
  float*    gden  = palloc((size_t)G);
  float*    gfeat = palloc((size_t)G * 384);
  float*    t1    = palloc((size_t)G * 256);
  float*    t2    = palloc((size_t)G * 128);
  float*    t3    = palloc((size_t)G * 64);
  float*    t4    = palloc((size_t)G * 64);
  float*    t5    = palloc((size_t)G * 32);

  // gate scores
  wmma_gemm_kernel<64, true, true><<<nblk(N, 128), 256, 0, stream>>>(hcur, gW1, gb1, g1, N, 128);
  rowdot_kernel<64, false><<<nblk(N, TB), TB, 0, stream>>>(g1, gW2, gb2, gatev, N);

  // pooling accumulators
  fill_f32<<<nblk((long long)G * 128, TB), TB, 0, stream>>>(gsum, 0.f, (long long)G * 128);
  fill_u32<<<nblk((long long)G * 128, TB), TB, 0, stream>>>(gmax, 0u, (long long)G * 128);
  fill_f32<<<nblk((long long)G * 128, TB), TB, 0, stream>>>(gatt, 0.f, (long long)G * 128);
  fill_f32<<<nblk(G, TB), TB, 0, stream>>>(gcnt, 0.f, G);
  fill_u32<<<nblk(G, TB), TB, 0, stream>>>(gm, 0u, G);
  fill_f32<<<nblk(G, TB), TB, 0, stream>>>(gden, 0.f, G);

  pool_acc_kernel<<<nblk((long long)N * 32, TB), TB, 0, stream>>>(hcur, gatev, batch, gsum, gmax, gcnt, gm, N);
  pool_exp_kernel<<<nblk(N, TB), TB, 0, stream>>>(gatev, batch, gm, ga, gden, N);
  pool_att_kernel<<<nblk((long long)N * 32, TB), TB, 0, stream>>>(hcur, ga, gden, batch, gatt, N);
  gfeat_kernel<<<nblk((long long)G * 384, TB), TB, 0, stream>>>(gsum, gmax, gatt, gcnt, gfeat, G);

  // ---- MLP head (all WMMA) ----
  wmma_gemm_kernel<256, true, true><<<nblk(G, 128), 256, 0, stream>>>(gfeat, mW1, mb1, t1, G, 384);
  wmma_gemm_kernel<128, true, true><<<nblk(G, 128), 256, 0, stream>>>(t1, mW2, mb2, t2, G, 256);
  wmma_gemm_kernel<64,  true, true><<<nblk(G, 128), 256, 0, stream>>>(t2, fW1, fb1, t3, G, 128);
  wmma_gemm_kernel<64,  true, true><<<nblk(G, 128), 256, 0, stream>>>(t3, fW2, fb2, t4, G, 64);
  wmma_gemm_kernel<32,  true, true><<<nblk(G, 128), 256, 0, stream>>>(t4, cW1, cb1, t5, G, 64);
  rowdot_kernel<32, true><<<nblk(G, TB), TB, 0, stream>>>(t5, cW2, cb2, (float*)d_out, G);
}
